// Net_4209067950300
// MI455X (gfx1250) — compile-verified, matrix operand order checked by reading, not run on hardware
//
#include <hip/hip_runtime.h>
#include <hip/hip_bf16.h>

// ---------------------------------------------------------------------------
// CDNA5 (gfx1250) PointNet++ segmentation pipeline.
// Matrix math runs on v_wmma_f32_16x16x32_f16 (wave32 WMMA).
// ---------------------------------------------------------------------------

typedef __attribute__((ext_vector_type(16))) _Float16 v16h;
typedef __attribute__((ext_vector_type(8)))  float    v8f;

union Frag {
    v16h    v;
    uint4   q[2];
    _Float16 h[16];
};

// ---------------------------------------------------------------------------
// Weight packing: W[K x N] f32 row-major -> per-(ktile,ntile) lane fragments
// matching the CDNA5 16-bit B-matrix VGPR layout (ISA 7.12.2):
//   lane l covers column n = nt*16 + (l%16)
//   lanes 0-15 cover k offsets {0..7, 16..23}; lanes 16-31 {8..15, 24..31}
// 16 halves per lane, stored contiguously (2 x b128 loads at use site).
// ---------------------------------------------------------------------------
__global__ void pack_weights(const float* __restrict__ W, _Float16* __restrict__ Wp,
                             int K, int N, int Kpad, int Npad) {
    int total = (Kpad >> 5) * (Npad >> 4) * 32;
    int g = blockIdx.x * blockDim.x + threadIdx.x;
    if (g >= total) return;
    int lane   = g & 31;
    int tile   = g >> 5;
    int ntiles = Npad >> 4;
    int nt = tile % ntiles, kt = tile / ntiles;
    int laneM = lane & 15, hi = lane >> 4;
    int n = nt * 16 + laneM;
    _Float16* dst = Wp + (size_t)g * 16;
#pragma unroll
    for (int h = 0; h < 16; ++h) {
        int dk = (h < 8 ? h : h + 8) + hi * 8;   // {0..7,16..23} or {8..15,24..31}
        int k  = kt * 32 + dk;
        float v = (k < K && n < N) ? W[(size_t)k * N + n] : 0.f;
        dst[h] = (_Float16)v;
    }
}

__global__ void f32_to_f16(const float* __restrict__ src, _Float16* __restrict__ dst, int total) {
    int g = blockIdx.x * blockDim.x + threadIdx.x;
    if (g < total) dst[g] = (_Float16)src[g];
}

// ---------------------------------------------------------------------------
// Farthest point sampling: one block per cloud, positions resident in LDS,
// sequential scan (n_sample-1 steps) with full-block argmax reduction.
// Tie-break = lowest index (matches jnp.argmax).
// ---------------------------------------------------------------------------
__global__ __launch_bounds__(1024)
void fps_kernel(const float* __restrict__ pos, int n, int ns, int* __restrict__ idx) {
    __shared__ float sxp[4096], syp[4096], szp[4096];
    __shared__ float rv[1024];
    __shared__ int   ri[1024];
    int b = blockIdx.x, t = threadIdx.x;
    const float* p = pos + (size_t)b * n * 3;
    for (int i = t; i < n; i += 1024) {
        sxp[i] = p[i * 3]; syp[i] = p[i * 3 + 1]; szp[i] = p[i * 3 + 2];
    }
    float mind[4];
#pragma unroll
    for (int j = 0; j < 4; ++j) mind[j] = 1e10f;
    if (t == 0) idx[(size_t)b * ns] = 0;
    int last = 0;
    __syncthreads();
    for (int s = 1; s < ns; ++s) {
        float lx = sxp[last], ly = syp[last], lz = szp[last];
        float bv = -1.f; int bi = 0x7fffffff;
#pragma unroll
        for (int j = 0; j < 4; ++j) {
            int i = j * 1024 + t;
            if (i < n) {
                float dx = sxp[i] - lx, dy = syp[i] - ly, dz = szp[i] - lz;
                float d  = dx * dx + dy * dy + dz * dz;
                float md = fminf(mind[j], d);
                mind[j] = md;
                if (md > bv) { bv = md; bi = i; }
            }
        }
        rv[t] = bv; ri[t] = bi;
        __syncthreads();
        for (int st = 512; st > 0; st >>= 1) {
            if (t < st) {
                float ov = rv[t + st]; int oi = ri[t + st];
                if (ov > rv[t] || (ov == rv[t] && oi < ri[t])) { rv[t] = ov; ri[t] = oi; }
            }
            __syncthreads();
        }
        last = ri[0];
        if (t == 0) idx[(size_t)b * ns + s] = last;
        __syncthreads();
    }
}

__global__ void gather3(const float* __restrict__ src, const int* __restrict__ idx,
                        float* __restrict__ dst, int S, int n, int total) {
    int g = blockIdx.x * blockDim.x + threadIdx.x;
    if (g >= total) return;
    int c = g % 3, r = (g / 3) % S, b = g / (3 * S);
    dst[g] = src[(size_t)b * n * 3 + (size_t)idx[(size_t)b * S + r] * 3 + c];
}

// ---------------------------------------------------------------------------
// 64-nearest-neighbor selection: one block per center, d^2 table in LDS,
// 64 rounds of block-wide min extraction (stable: lowest index on ties).
// ---------------------------------------------------------------------------
__global__ __launch_bounds__(256)
void topk64(const float* __restrict__ pos, const float* __restrict__ centers,
            int n, int S, int* __restrict__ nbr, float* __restrict__ d2o) {
    __shared__ float sd[4096];
    __shared__ float rv[256];
    __shared__ int   ri[256];
    int b = blockIdx.x / S;
    int t = threadIdx.x;
    const float* p = pos + (size_t)b * n * 3;
    const float* c = centers + (size_t)blockIdx.x * 3;
    float cx = c[0], cy = c[1], cz = c[2];
    for (int i = t; i < n; i += 256) {
        float dx = p[i * 3] - cx, dy = p[i * 3 + 1] - cy, dz = p[i * 3 + 2] - cz;
        sd[i] = dx * dx + dy * dy + dz * dz;
    }
    __syncthreads();
    int*   onb = nbr + (size_t)blockIdx.x * 64;
    float* od  = d2o + (size_t)blockIdx.x * 64;
    for (int r = 0; r < 64; ++r) {
        float bv = 3e38f; int bi = 0x7fffffff;
        for (int i = t; i < n; i += 256) {
            float v = sd[i];
            if (v < bv || (v == bv && i < bi)) { bv = v; bi = i; }
        }
        rv[t] = bv; ri[t] = bi;
        __syncthreads();
        for (int st = 128; st > 0; st >>= 1) {
            if (t < st) {
                float ov = rv[t + st]; int oi = ri[t + st];
                if (ov < rv[t] || (ov == rv[t] && oi < ri[t])) { rv[t] = ov; ri[t] = oi; }
            }
            __syncthreads();
        }
        if (t == 0) { onb[r] = ri[0]; od[r] = rv[0]; sd[ri[0]] = 3e38f; }
        __syncthreads();
    }
}

// ---------------------------------------------------------------------------
// One WMMA MLP layer over a 64-row tile held in LDS.
// 8 waves = 4 M-tiles x 2 N-wavegroups; each wave owns 16x16 tiles.
// A-fragment layout per ISA 7.12.2 (16-bit A 16x32): lane row = lane%16,
// lanes 0-15 k {0..7,16..23}, lanes 16-31 k {8..15,24..31} -> two b128 loads.
// ---------------------------------------------------------------------------
__device__ inline void mlp_layer(const _Float16* __restrict__ src, int lds, int K,
                                 _Float16* __restrict__ dst, int ldd, int Nl,
                                 const _Float16* __restrict__ Wp,
                                 const float* __restrict__ bias, int tid) {
    int wave = tid >> 5, lane = tid & 31;
    int mt = wave & 3, nw = wave >> 2;
    int laneM = lane & 15, hi = lane >> 4;
    int row = mt * 16 + laneM;
    int ntiles = Nl >> 4;
    for (int nt = nw; nt < ntiles; nt += 2) {
        v8f acc = {};
        for (int kt = 0; kt < (K >> 5); ++kt) {
            Frag a, bfr;
            const _Float16* ap = src + row * lds + kt * 32 + hi * 8;
            a.q[0] = *(const uint4*)ap;
            a.q[1] = *(const uint4*)(ap + 16);
            const _Float16* bp = Wp + ((size_t)(kt * ntiles + nt) * 32 + lane) * 16;
            bfr.q[0] = *(const uint4*)bp;
            bfr.q[1] = *(const uint4*)(bp + 8);
            acc = __builtin_amdgcn_wmma_f32_16x16x32_f16(
                false, a.v, false, bfr.v, (short)0, acc, false, false);
        }
        int n = nt * 16 + laneM;
        float bv = bias[n];
#pragma unroll
        for (int r = 0; r < 8; ++r) {
            int m = mt * 16 + r + hi * 8;      // C layout: lanes16-31 hold M=r+8
            float v = fmaxf(acc[r] + bv, 0.f); // SA MLP layers are all ReLU
            dst[m * ldd + n] = (_Float16)v;
        }
    }
}

// ---------------------------------------------------------------------------
// Fused SA module: one block per sampled center (64 neighbors == one 64-row
// WMMA tile). Gather features -> 3 WMMA layers in LDS -> masked max-pool.
// ---------------------------------------------------------------------------
__global__ __launch_bounds__(256)
void fused_sa(const _Float16* __restrict__ xin, int sx, int Cx, int n,
              const float* __restrict__ pos, const float* __restrict__ centers,
              const int* __restrict__ nbr, const float* __restrict__ d2sel, float r2,
              const _Float16* __restrict__ W1, const float* __restrict__ b1,
              const _Float16* __restrict__ W2, const float* __restrict__ b2,
              const _Float16* __restrict__ W3, const float* __restrict__ b3,
              int Kpad, int H1, int H2, int H3,
              _Float16* __restrict__ xout, int S) {
    __shared__ _Float16 X[64 * 160];   // ping: feat (K<=160) / hidden2 (<=128)
    __shared__ _Float16 Y[64 * 256];   // pong: hidden1 (<=128) / hidden3 (<=256)
    __shared__ float sd2[64];

    int b = blockIdx.x / S;
    int t = threadIdx.x;
    const _Float16* xc = xin + (size_t)b * n * sx;
    const float* pc = pos + (size_t)b * n * 3;
    const float* ct = centers + (size_t)blockIdx.x * 3;
    const int*   nb = nbr + (size_t)blockIdx.x * 64;
    const float* dd = d2sel + (size_t)blockIdx.x * 64;
    float c0 = ct[0], c1 = ct[1], c2 = ct[2];
    if (t < 64) sd2[t] = dd[t];

    // Gather [x_j, pos_j - center] into the 64 x Kpad f16 tile (zero padded).
    for (int i = t; i < 64 * Kpad; i += 256) {
        int row = i / Kpad, c = i - row * Kpad;
        int j = nb[row];
        float v;
        if (c < Cx)          v = (float)xc[(size_t)j * sx + c];
        else if (c < Cx + 3) { int d = c - Cx;
                               v = pc[(size_t)j * 3 + d] - (d == 0 ? c0 : (d == 1 ? c1 : c2)); }
        else                 v = 0.f;
        X[row * 160 + c] = (_Float16)v;
    }
    __syncthreads();
    mlp_layer(X, 160, Kpad, Y, 256, H1, W1, b1, t);  __syncthreads();
    mlp_layer(Y, 256, H1,   X, 160, H2, W2, b2, t);  __syncthreads();
    mlp_layer(X, 160, H2,   Y, 256, H3, W3, b3, t);  __syncthreads();

    // Masked max over the 64 neighbors (invalid -> -1e9 as in reference).
    for (int c = t; c < H3; c += 256) {
        float mx = -1e9f;
#pragma unroll 4
        for (int k = 0; k < 64; ++k) {
            float v = (sd2[k] <= r2) ? (float)Y[k * 256 + c] : -1e9f;
            mx = fmaxf(mx, v);
        }
        xout[(size_t)blockIdx.x * H3 + c] = (_Float16)mx;
    }
}

// ---------------------------------------------------------------------------
// Generic WMMA GEMM + bias + (optional) ReLU:  C[M x ldc] = act(A x W + b)
// A: f16 row-major (lda = Kpad, mult of 32, zero-padded), Wp: packed frags.
// Block tile 64(M) x 32(N); 8 waves = 4M x 2N of 16x16 WMMA tiles.
// ---------------------------------------------------------------------------
__global__ __launch_bounds__(256)
void gemm_bias_act(const _Float16* __restrict__ A, int lda,
                   const _Float16* __restrict__ Wp,
                   const float* __restrict__ bias,
                   _Float16* __restrict__ C, int ldc,
                   int M, int K, int N, int relu) {
    __shared__ _Float16 At[64 * 32];
    int t = threadIdx.x;
    int wave = t >> 5, lane = t & 31;
    int mt = wave & 3, nw = wave >> 2;
    int laneM = lane & 15, hi = lane >> 4;
    int m0 = blockIdx.x * 64;
    int ntAbs = blockIdx.y * 2 + nw;
    int n = ntAbs * 16 + laneM;
    int ntilesTot = ldc >> 4;          // Npad / 16
    int lrow = t >> 2, lseg = t & 3;   // cooperative A-tile load mapping
    v8f acc = {};
    for (int kt = 0; kt < (K >> 5); ++kt) {
        __syncthreads();
        uint4 val = {0u, 0u, 0u, 0u};
        int gm = m0 + lrow;
        if (gm < M) val = *(const uint4*)(A + (size_t)gm * lda + kt * 32 + lseg * 8);
        *(uint4*)(At + lrow * 32 + lseg * 8) = val;
        __syncthreads();
        Frag a, bfr;
        const _Float16* ap = At + (mt * 16 + laneM) * 32 + hi * 8;
        a.q[0] = *(const uint4*)ap;
        a.q[1] = *(const uint4*)(ap + 16);
        const _Float16* bp = Wp + ((size_t)(kt * ntilesTot + ntAbs) * 32 + lane) * 16;
        __builtin_prefetch(bp + (size_t)ntilesTot * 32 * 16, 0, 1); // next k-tile of W
        bfr.q[0] = *(const uint4*)bp;
        bfr.q[1] = *(const uint4*)(bp + 8);
        acc = __builtin_amdgcn_wmma_f32_16x16x32_f16(
            false, a.v, false, bfr.v, (short)0, acc, false, false);
    }
    float bv = (n < N) ? bias[n] : 0.f;
#pragma unroll
    for (int r = 0; r < 8; ++r) {
        int m = m0 + mt * 16 + r + hi * 8;
        if (m < M) {
            float v = acc[r] + bv;
            if (relu) v = fmaxf(v, 0.f);
            C[(size_t)m * ldc + n] = (n < N) ? (_Float16)v : (_Float16)0.f;
        }
    }
}

// ---------------------------------------------------------------------------
// Misc glue kernels
// ---------------------------------------------------------------------------
__global__ void assemble_sa3(const _Float16* __restrict__ x2, const float* __restrict__ pos2,
                             _Float16* __restrict__ A, int rows) {
    int g = blockIdx.x * blockDim.x + threadIdx.x;
    if (g >= rows * 288) return;
    int m = g / 288, c = g - m * 288;
    _Float16 v;
    if (c < 256)      v = x2[(size_t)m * 256 + c];
    else if (c < 259) v = (_Float16)pos2[(size_t)m * 3 + (c - 256)];
    else              v = (_Float16)0.f;
    A[g] = v;
}

__global__ void colmax(const _Float16* __restrict__ G, _Float16* __restrict__ xg,
                       int rowsPerCloud, int C, int total) {
    int g = blockIdx.x * blockDim.x + threadIdx.x;
    if (g >= total) return;
    int b = g / C, c = g % C;
    float mx = -3e38f;
    const _Float16* base = G + (size_t)b * rowsPerCloud * C + c;
    for (int k = 0; k < rowsPerCloud; ++k) mx = fmaxf(mx, (float)base[(size_t)k * C]);
    xg[g] = (_Float16)mx;
}

__global__ void assemble_fp3(const _Float16* __restrict__ xg, const _Float16* __restrict__ x2,
                             _Float16* __restrict__ A, int rowsPerCloud, int total) {
    int g = blockIdx.x * blockDim.x + threadIdx.x;
    if (g >= total) return;
    int m = g / 1280, c = g - m * 1280;
    int b = m / rowsPerCloud;
    A[g] = (c < 1024) ? xg[(size_t)b * 1024 + c] : x2[(size_t)m * 256 + (c - 1024)];
}

// 3-NN inverse-squared-distance interpolation fused with skip-concat.
__global__ void interp3_concat(const float* __restrict__ posDst, const float* __restrict__ posSrc,
                               const _Float16* __restrict__ fsrc, const _Float16* __restrict__ fskip,
                               _Float16* __restrict__ A, int M, int S, int Cs, int Ck, int Kpad) {
    int g = blockIdx.x * blockDim.x + threadIdx.x;
    if (g >= 4 * M) return;
    int b = g / M;
    const float* pd = posDst + (size_t)g * 3;
    const float* ps = posSrc + (size_t)b * S * 3;
    float px = pd[0], py = pd[1], pz = pd[2];
    float d0 = 3e38f, d1 = 3e38f, d2 = 3e38f;
    int   i0 = 0, i1 = 0, i2 = 0;
    for (int i = 0; i < S; ++i) {
        float ax = ps[i * 3] - px, ay = ps[i * 3 + 1] - py, az = ps[i * 3 + 2] - pz;
        float v = ax * ax + ay * ay + az * az;
        if (v < d0)      { d2 = d1; i2 = i1; d1 = d0; i1 = i0; d0 = v; i0 = i; }
        else if (v < d1) { d2 = d1; i2 = i1; d1 = v; i1 = i; }
        else if (v < d2) { d2 = v; i2 = i; }
    }
    float w0 = 1.f / fmaxf(d0, 1e-16f);
    float w1 = 1.f / fmaxf(d1, 1e-16f);
    float w2 = 1.f / fmaxf(d2, 1e-16f);
    float ws = w0 + w1 + w2;
    w0 /= ws; w1 /= ws; w2 /= ws;
    const _Float16* f0 = fsrc + ((size_t)b * S + i0) * Cs;
    const _Float16* f1 = fsrc + ((size_t)b * S + i1) * Cs;
    const _Float16* f2 = fsrc + ((size_t)b * S + i2) * Cs;
    _Float16* row = A + (size_t)g * Kpad;
    for (int c = 0; c < Cs; ++c)
        row[c] = (_Float16)(w0 * (float)f0[c] + w1 * (float)f1[c] + w2 * (float)f2[c]);
    const _Float16* sk = fskip + (size_t)g * Ck;
    for (int c = 0; c < Ck; ++c) row[Cs + c] = sk[c];
    for (int c = Cs + Ck; c < Kpad; ++c) row[c] = (_Float16)0.f;
}

__global__ void sigmoid3(const _Float16* __restrict__ L, float* __restrict__ out, int rows) {
    int g = blockIdx.x * blockDim.x + threadIdx.x;
    if (g >= rows * 3) return;
    int m = g / 3, j = g % 3;
    float v = (float)L[(size_t)m * 32 + j];
    out[g] = 1.f / (1.f + expf(-v));
}

// ---------------------------------------------------------------------------
// Host orchestration
// ---------------------------------------------------------------------------
extern "C" void kernel_launch(void* const* d_in, const int* in_sizes, int n_in,
                              void* d_out, int out_size, void* d_ws, size_t ws_size,
                              hipStream_t stream) {
    (void)in_sizes; (void)n_in; (void)out_size; (void)ws_size;
    const int Bc = 4, Nn = 4096, S1 = 2048, S2 = 1024;
    const float* x32 = (const float*)d_in[0];
    const float* pos = (const float*)d_in[1];
    auto Wf = [&](int i) { return (const float*)d_in[i]; };

    size_t off = 0;
    auto alloc  = [&](size_t bytes) -> char* {
        size_t a = (off + 255) & ~(size_t)255;
        off = a + bytes;
        return (char*)d_ws + a;
    };
    auto allocH = [&](size_t n) -> _Float16* { return (_Float16*)alloc(n * 2); };
    auto allocF = [&](size_t n) -> float*    { return (float*)alloc(n * 4); };
    auto allocI = [&](size_t n) -> int*      { return (int*)alloc(n * 4); };

    // Weight packing descriptors: d_in index of W (bias = idx+1), K, N, Kpad, Npad.
    struct L { int wi; int K, N, Kp, Np; _Float16* wp; };
    L Ls[19] = {
        {3, 6, 64, 32, 64, nullptr},   {5, 64, 64, 64, 64, nullptr},   {7, 64, 128, 64, 128, nullptr},
        {9, 131, 128, 160, 128, nullptr},{11,128,128,128,128, nullptr},{13,128,256,128,256, nullptr},
        {15,259,256,288,256, nullptr}, {17,256,512,256,512, nullptr}, {19,512,1024,512,1024, nullptr},
        {21,1280,256,1280,256, nullptr},{23,256,256,256,256, nullptr},
        {25,384,256,384,256, nullptr}, {27,256,128,256,128, nullptr},
        {29,131,128,160,128, nullptr}, {31,128,128,128,128, nullptr}, {33,128,128,128,128, nullptr},
        {35,128,128,128,128, nullptr}, {37,128,128,128,128, nullptr}, {39,128,3,128,32, nullptr}
    };
    for (auto& l : Ls) l.wp = allocH((size_t)l.Kp * l.Np);

    _Float16* xh   = allocH((size_t)Bc * Nn * 3);
    int*      idx1 = allocI((size_t)Bc * S1);
    float*    ctr1 = allocF((size_t)Bc * S1 * 3);
    int*      nbr1 = allocI((size_t)Bc * S1 * 64);
    float*    d21  = allocF((size_t)Bc * S1 * 64);
    _Float16* x1h  = allocH((size_t)Bc * S1 * 128);
    int*      idx2 = allocI((size_t)Bc * S2);
    float*    pos2 = allocF((size_t)Bc * S2 * 3);
    int*      nbr2 = allocI((size_t)Bc * S2 * 64);
    float*    d22  = allocF((size_t)Bc * S2 * 64);
    _Float16* x2h  = allocH((size_t)Bc * S2 * 256);
    _Float16* A3   = allocH((size_t)Bc * S2 * 288);
    _Float16* G1   = allocH((size_t)Bc * S2 * 256);
    _Float16* G2   = allocH((size_t)Bc * S2 * 512);
    _Float16* G3   = allocH((size_t)Bc * S2 * 1024);
    _Float16* xg   = allocH((size_t)Bc * 1024);
    _Float16* A4   = allocH((size_t)Bc * S2 * 1280);
    _Float16* T1   = allocH((size_t)Bc * S2 * 256);
    _Float16* f3   = allocH((size_t)Bc * S2 * 256);
    _Float16* A5   = allocH((size_t)Bc * S1 * 384);
    _Float16* T2   = allocH((size_t)Bc * S1 * 256);
    _Float16* f2   = allocH((size_t)Bc * S1 * 128);
    _Float16* A6   = allocH((size_t)Bc * Nn * 160);
    _Float16* T3   = allocH((size_t)Bc * Nn * 128);
    _Float16* T4   = allocH((size_t)Bc * Nn * 128);
    _Float16* f1   = allocH((size_t)Bc * Nn * 128);
    _Float16* L1o  = allocH((size_t)Bc * Nn * 128);
    _Float16* L2o  = allocH((size_t)Bc * Nn * 128);
    _Float16* L3o  = allocH((size_t)Bc * Nn * 32);

    auto gemm = [&](const _Float16* A, int lda, const L& l, _Float16* C, int ldc,
                    int M, int relu) {
        dim3 grid((M + 63) / 64, ldc / 32);
        gemm_bias_act<<<grid, 256, 0, stream>>>(A, lda, l.wp, Wf(l.wi + 1), C, ldc,
                                                M, l.Kp, l.N, relu);
    };

    // 0) pack weights + convert x to f16
    for (auto& l : Ls) {
        int total = (l.Kp / 32) * (l.Np / 16) * 32;
        pack_weights<<<(total + 255) / 256, 256, 0, stream>>>(Wf(l.wi), l.wp,
                                                              l.K, l.N, l.Kp, l.Np);
    }
    {
        int total = Bc * Nn * 3;
        f32_to_f16<<<(total + 255) / 256, 256, 0, stream>>>(x32, xh, total);
    }

    // 1) SA1: fps -> 64-NN -> fused WMMA MLP + masked max
    fps_kernel<<<Bc, 1024, 0, stream>>>(pos, Nn, S1, idx1);
    gather3<<<(Bc * S1 * 3 + 255) / 256, 256, 0, stream>>>(pos, idx1, ctr1, S1, Nn, Bc * S1 * 3);
    topk64<<<Bc * S1, 256, 0, stream>>>(pos, ctr1, Nn, S1, nbr1, d21);
    fused_sa<<<Bc * S1, 256, 0, stream>>>(xh, 3, 3, Nn, pos, ctr1, nbr1, d21, 0.01f,
                                          Ls[0].wp, Wf(4), Ls[1].wp, Wf(6), Ls[2].wp, Wf(8),
                                          32, 64, 64, 128, x1h, S1);

    // 2) SA2
    fps_kernel<<<Bc, 1024, 0, stream>>>(ctr1, S1, S2, idx2);
    gather3<<<(Bc * S2 * 3 + 255) / 256, 256, 0, stream>>>(ctr1, idx2, pos2, S2, S1, Bc * S2 * 3);
    topk64<<<Bc * S2, 256, 0, stream>>>(ctr1, pos2, S1, S2, nbr2, d22);
    fused_sa<<<Bc * S2, 256, 0, stream>>>(x1h, 128, 128, S1, ctr1, pos2, nbr2, d22, 0.25f,
                                          Ls[3].wp, Wf(10), Ls[4].wp, Wf(12), Ls[5].wp, Wf(14),
                                          160, 128, 128, 256, x2h, S2);

    // 3) Global SA: mlp([x2, pos2]) then per-cloud max pool
    assemble_sa3<<<(Bc * S2 * 288 + 255) / 256, 256, 0, stream>>>(x2h, pos2, A3, Bc * S2);
    gemm(A3, 288, Ls[6], G1, 256, Bc * S2, 1);
    gemm(G1, 256, Ls[7], G2, 512, Bc * S2, 1);
    gemm(G2, 512, Ls[8], G3, 1024, Bc * S2, 1);
    colmax<<<(Bc * 1024 + 255) / 256, 256, 0, stream>>>(G3, xg, S2, 1024, Bc * 1024);

    // 4) FP3: 1-NN interp from global vector == broadcast; concat x2
    assemble_fp3<<<(Bc * S2 * 1280 + 255) / 256, 256, 0, stream>>>(xg, x2h, A4, S2,
                                                                   Bc * S2 * 1280);
    gemm(A4, 1280, Ls[9], T1, 256, Bc * S2, 1);
    gemm(T1, 256, Ls[10], f3, 256, Bc * S2, 1);

    // 5) FP2: 3-NN interp pos2->pos1, concat x1
    interp3_concat<<<(Bc * S1 + 255) / 256, 256, 0, stream>>>(ctr1, pos2, f3, x1h, A5,
                                                              S1, S2, 256, 128, 384);
    gemm(A5, 384, Ls[11], T2, 256, Bc * S1, 1);
    gemm(T2, 256, Ls[12], f2, 128, Bc * S1, 1);

    // 6) FP1: 3-NN interp pos1->pos, concat x
    interp3_concat<<<(Bc * Nn + 255) / 256, 256, 0, stream>>>(pos, ctr1, f2, xh, A6,
                                                              Nn, S1, 128, 3, 160);
    gemm(A6, 160, Ls[13], T3, 128, Bc * Nn, 1);
    gemm(T3, 128, Ls[14], T4, 128, Bc * Nn, 1);
    gemm(T4, 128, Ls[15], f1, 128, Bc * Nn, 1);

    // 7) Head: lin1(relu) -> lin2 -> lin3 -> sigmoid
    gemm(f1, 128, Ls[16], L1o, 128, Bc * Nn, 1);
    gemm(L1o, 128, Ls[17], L2o, 128, Bc * Nn, 0);
    gemm(L2o, 128, Ls[18], L3o, 32, Bc * Nn, 0);
    sigmoid3<<<(Bc * Nn * 3 + 255) / 256, 256, 0, stream>>>(L3o, (float*)d_out, Bc * Nn);
}
